// TeluCodec_55284819034253
// MI455X (gfx1250) — matrix operand order repaired and checked by reading, not compiled, not run on hardware
//
#include <hip/hip_runtime.h>
#include <hip/hip_bf16.h>
#include <math.h>

// ---------------------------------------------------------------------------
// CDNA5 (gfx1250) neural audio codec forward pass.
// Channel mixing (conv1d / conv-transpose1d / VQ distances) runs through
// v_wmma_f32_16x16x32_bf16 (wave32, f32 accumulate). VQ residual tiles are
// staged into LDS by the Tensor Data Mover (tensor_load_to_lds + tensorcnt).
// ---------------------------------------------------------------------------

typedef __attribute__((ext_vector_type(16))) __bf16 v16bf;
typedef __attribute__((ext_vector_type(8)))  float  v8f;
typedef __attribute__((ext_vector_type(4)))  unsigned int uint32x4;
typedef __attribute__((ext_vector_type(8)))  int   int32x8;
typedef __attribute__((ext_vector_type(4)))  int   int32x4;

// ============================================================================
// Implicit-GEMM conv1d / convT1d, specialized on (KW, STRIDE, MODE) so all
// index div/mod is by compile-time constants; fragment gathers use incremental
// ci/kw/position tracking (no per-element division).
//   MODE 0: y[b,co,o] = bias[co] + sum_{ci,kw} x[b,ci,o*ST+kw-padL] * w[co,ci,kw]
//   MODE 1 (ConvTranspose, weights (Cin,Cout,KW)):
//           y[b,co,o] = bias[co] + sum xup[o+kw-(KW-1-padL)] * w[ci,co,KW-1-kw]
// Tile: M=16 out-channels x N=128 positions per 256-thread block (8 waves).
// Weight tile staged in LDS and shared by all 8 waves.
// ============================================================================
template <int KW, int ST, int MODE>
__global__ void conv1d_wmma_k(const float* __restrict__ x,
                              const float* __restrict__ w,
                              const float* __restrict__ bias,
                              float* __restrict__ y,
                              int Cin, int Cout, int padL, int Lin, int Lout)
{
  __shared__ __bf16 ldsA[16][40];          // 16x32 weight tile, padded rows
  const int tid  = threadIdx.x;
  const int lane = tid & 31;
  const int wave = tid >> 5;
  const int half = lane >> 4;              // WMMA lane-half
  const int mrow = lane & 15;
  const int ncol = lane & 15;
  const int co0  = blockIdx.y * 16;
  const int l0   = blockIdx.x * 128 + wave * 16;
  const int b    = blockIdx.z;
  const int Ktot = Cin * KW;

  const float* __restrict__ xb = x + (long)b * Cin * Lin;
  v8f acc = {};

  const int pos_out = l0 + ncol;

  for (int kk = 0; kk < Ktot; kk += 32) {
    // ---- cooperative A (weight) tile load: 16 rows x 32 k ----
    for (int e = tid; e < 16 * 32; e += 256) {
      int m = e >> 5, k = e & 31;
      int gk = kk + k, co = co0 + m;
      float v = 0.f;
      if (co < Cout && gk < Ktot) {
        if (MODE == 0) {
          v = w[(long)co * Ktot + gk];
        } else {
          int ci = gk / KW, kwi = gk - ci * KW;     // constant divisor
          v = w[((long)ci * Cout + co) * KW + (KW - 1 - kwi)];
        }
      }
      ldsA[m][k] = (__bf16)v;
    }
    __syncthreads();

    // ---- A fragment from LDS (16-bit A 16x32 layout) ----
    v16bf afrag;
#pragma unroll
    for (int i = 0; i < 8; ++i) {
      afrag[i]     = ldsA[mrow][half * 16 + i];
      afrag[8 + i] = ldsA[mrow][half * 16 + 8 + i];
    }

    if (kk + 32 < Ktot)
      __builtin_prefetch(xb + (long)((kk + 32) / KW) * Lin +
                         (MODE == 0 ? pos_out * ST : pos_out), 0, 1);

    // ---- B fragment: incremental im2col gather ----
    const int gk0 = kk + half * 16;
    int ci  = gk0 / KW;                                // constant divisor
    int kwi = gk0 - ci * KW;
    v16bf bfrag;
    if (MODE == 0) {
      const int pbase = pos_out * ST - padL;
      int p = pbase + kwi;
#pragma unroll
      for (int t = 0; t < 16; ++t) {
        float v = 0.f;
        if (ci < Cin && p >= 0 && p < Lin) v = xb[(long)ci * Lin + p];
        bfrag[t] = (__bf16)v;
        ++kwi; ++p;
        if (kwi == KW) { kwi = 0; ++ci; p = pbase; }
      }
    } else {
      const int pbase = pos_out - (KW - 1 - padL);
      // floor-div decomposition of pbase and of p0 = pbase + kwi by ST
      int qB  = (pbase >= 0) ? pbase / ST : -((-pbase + ST - 1) / ST);
      int pmB = pbase - qB * ST;
      int p   = pbase + kwi;
      int qv  = (p >= 0) ? p / ST : -((-p + ST - 1) / ST);
      int pm  = p - qv * ST;
#pragma unroll
      for (int t = 0; t < 16; ++t) {
        float v = 0.f;
        if (ci < Cin && p >= 0 && pm == 0 && qv >= 0 && qv < Lin)
          v = xb[(long)ci * Lin + qv];
        bfrag[t] = (__bf16)v;
        ++kwi; ++p; ++pm;
        if (pm == ST) { pm = 0; ++qv; }
        if (kwi == KW) { kwi = 0; ++ci; p = pbase; pm = pmB; qv = qB; }
      }
    }

    acc = __builtin_amdgcn_wmma_f32_16x16x32_bf16(
        false, afrag, false, bfrag, (short)0, acc, false, false);
    __syncthreads();
  }

  // ---- epilogue: D layout (VGPR r -> M = r + half*8, N = lane&15) ----
#pragma unroll
  for (int r = 0; r < 8; ++r) {
    int m  = r + half * 8;
    int co = co0 + m;
    int po = l0 + ncol;
    if (co < Cout && po < Lout)
      y[((long)b * Cout + co) * Lout + po] = acc[r] + bias[co];
  }
}

// ============================================================================
// GroupNorm(1, C) statistics: per-sample mean / rsqrt(var) over (C, L).
// ============================================================================
__global__ void gn_stats_k(const float* __restrict__ x, float* __restrict__ stats,
                           int C, long L)
{
  int b = blockIdx.x;
  long n = (long)C * L;
  const float* xb = x + (long)b * n;
  double s = 0.0, ss = 0.0;
  for (long i = threadIdx.x; i < n; i += blockDim.x) {
    float v = xb[i];
    s += v; ss += (double)v * v;
  }
  __shared__ double sh0[256], sh1[256];
  sh0[threadIdx.x] = s; sh1[threadIdx.x] = ss;
  __syncthreads();
  for (int o = 128; o > 0; o >>= 1) {
    if (threadIdx.x < o) { sh0[threadIdx.x] += sh0[threadIdx.x + o];
                           sh1[threadIdx.x] += sh1[threadIdx.x + o]; }
    __syncthreads();
  }
  if (threadIdx.x == 0) {
    double m   = sh0[0] / (double)n;
    double var = sh1[0] / (double)n - m * m;
    stats[2 * b]     = (float)m;
    stats[2 * b + 1] = rsqrtf((float)var + 1e-5f);
  }
}

// fused GroupNorm-apply + snake activation: v = gn(x); y = v + sin(a v)^2 / a
__global__ void gn_snake_k(const float* __restrict__ x, float* __restrict__ y,
                           const float* __restrict__ stats,
                           const float* __restrict__ g, const float* __restrict__ bb,
                           const float* __restrict__ al, int C, long L, long n)
{
  long i = (long)blockIdx.x * blockDim.x + threadIdx.x;
  if (i >= n) return;
  long CL = (long)C * L;
  int b = (int)(i / CL);
  int c = (int)((i / L) % C);
  float m = stats[2 * b], rs = stats[2 * b + 1];
  float v = (x[i] - m) * rs * g[c] + bb[c];
  float a = al[c];
  float s = sinf(a * v);
  y[i] = v + s * s / a;
}

__global__ void add_inplace_k(float* __restrict__ x, const float* __restrict__ r, long n)
{
  long i = (long)blockIdx.x * blockDim.x + threadIdx.x;
  if (i < n) x[i] += r[i];
}

__global__ void fill_zero_k(float* __restrict__ p, long n)
{
  long i = (long)blockIdx.x * blockDim.x + threadIdx.x;
  if (i < n) p[i] = 0.f;
}

// (B,D,T) -> token-major (B*T, D)
__global__ void transpose_dt_k(const float* __restrict__ z, float* __restrict__ o,
                               int B, int D, int T)
{
  long i = (long)blockIdx.x * blockDim.x + threadIdx.x;
  long n = (long)B * D * T;
  if (i >= n) return;
  int t = (int)(i % T);
  int d = (int)((i / T) % D);
  int b = (int)(i / ((long)D * T));
  o[((long)b * T + t) * D + d] = z[i];
}

// token-major (B*T, D) -> (B,D,T)
__global__ void transpose_td_k(const float* __restrict__ q, float* __restrict__ z,
                               int B, int D, int T)
{
  long i = (long)blockIdx.x * blockDim.x + threadIdx.x;
  long n = (long)B * D * T;
  if (i >= n) return;
  int t = (int)(i % T);
  int d = (int)((i / T) % D);
  int b = (int)(i / ((long)D * T));
  z[i] = q[((long)b * T + t) * D + d];
}

// per-code squared norms, all 8 codebooks at once
__global__ void csq_k(const float* __restrict__ cb, float* __restrict__ csq,
                      int ncodes, int D)
{
  int i = blockIdx.x * blockDim.x + threadIdx.x;
  if (i >= ncodes) return;
  const float* r = cb + (long)i * D;
  float s = 0.f;
  for (int d = 0; d < D; ++d) { float v = r[d]; s += v * v; }
  csq[i] = s;
}

// ============================================================================
// VQ argmin via WMMA: one wave handles 16 tokens.
// Residual tile (16 x 1024 fp32 = 64KB) is DMA'd into LDS by the Tensor Data
// Mover (tensor_load_to_lds, TENSORcnt), converted once to a bf16 tile, then
// swept against 64 code tiles with WMMA; score = csq[n] - 2*dot.
// Codebook (4MB/quantizer) stays resident in the 192MB L2 across all blocks.
// ============================================================================
__global__ void vq_argmin_k(const float* __restrict__ resid,
                            const float* __restrict__ cb,
                            const float* __restrict__ csq,
                            float* __restrict__ codes_out,  // (B,NQ,T) as float
                            int* __restrict__ idx_out,
                            int tokens, int T, int q, int NQ)
{
  // single LDS arena so the TDM destination sits at static-LDS offset 0
  __shared__ __align__(16) unsigned char smem[16 * 1024 * 4 + 16 * 1040 * 2];
  float* rtf = (float*)smem;                                   // fp32 stage (TDM)
  __bf16 (*rt)[1040] = (__bf16 (*)[1040])(smem + 16 * 1024 * 4); // bf16 compute tile

  const int lane = threadIdx.x & 31;
  const int half = lane >> 4;
  const int mrow = lane & 15;
  const int ncol = lane & 15;
  const int tok0 = blockIdx.x * 16;
  const int D = 1024;

  int rows = tokens - tok0;
  if (rows > 16) rows = 16;

  // ---- Tensor DMA descriptor (D#): 2D tile 1024 x rows of 4B elements ----
  unsigned long long ga = (unsigned long long)(const void*)(resid + (long)tok0 * D);
  uint32x4 g0;
  g0[0] = 1u;                                            // count=1, user mode
  g0[1] = 0u;                                            // lds_addr = 0
  g0[2] = (unsigned)(ga & 0xFFFFFFFFu);                  // global_addr[31:0]
  g0[3] = (unsigned)((ga >> 32) & 0x01FFFFFFu) | (2u << 30); // addr[56:32] | type=2
  int32x8 g1;
  g1[0] = (int)(2u << 16);                // data_size=4B, no flags, wg_mask=0
  g1[1] = (int)(1024u << 16);             // tensor_dim0[15:0]=1024 (bits 79:48 lo)
  g1[2] = (int)((unsigned)rows << 16);    // tensor_dim0 hi=0 | tensor_dim1=rows
  g1[3] = (int)(1024u << 16);             // tensor_dim1 hi=0 | tile_dim0=1024
  g1[4] = 16;                             // tile_dim1=16, tile_dim2=0
  g1[5] = 1024;                           // tensor_dim0_stride[31:0]=1024
  g1[6] = 0;                              // stride hi, tensor_dim1_stride lo
  g1[7] = 0;
  int32x4 gz4 = {0, 0, 0, 0};
  int32x8 gz8 = {0, 0, 0, 0, 0, 0, 0, 0};
  // amdgpu-toolchain (clang-23) 6-arg form: (g0, g1, g2, g3, g4, cpol)
  __builtin_amdgcn_tensor_load_to_lds(g0, g1, gz4, gz4, gz8, 0);
  __builtin_amdgcn_s_wait_tensorcnt((short)0);
  __syncthreads();

  // one-time fp32 -> bf16 conversion into the padded compute tile
  for (int i = lane; i < 16 * 1024; i += 32) {
    int m = i >> 10, d = i & 1023;
    rt[m][d] = (__bf16)rtf[i];
  }
  __syncthreads();

  float bestv[8];
  int   besti[8];
#pragma unroll
  for (int r = 0; r < 8; ++r) { bestv[r] = 3.4e38f; besti[r] = 0; }

  for (int n0 = 0; n0 < 1024; n0 += 16) {
    v8f acc = {};
    for (int kk = 0; kk < D; kk += 32) {
      v16bf afrag;
#pragma unroll
      for (int i = 0; i < 8; ++i) {
        afrag[i]     = rt[mrow][kk + half * 16 + i];
        afrag[8 + i] = rt[mrow][kk + half * 16 + 8 + i];
      }
      const float* crow = cb + (long)(n0 + ncol) * D + kk + half * 16;
      v16bf bfrag;
#pragma unroll
      for (int t = 0; t < 16; ++t) bfrag[t] = (__bf16)crow[t];
      acc = __builtin_amdgcn_wmma_f32_16x16x32_bf16(
          false, afrag, false, bfrag, (short)0, acc, false, false);
    }
#pragma unroll
    for (int r = 0; r < 8; ++r) {
      int n = n0 + ncol;
      float sc = csq[n] - 2.f * acc[r];
      if (sc < bestv[r] || (sc == bestv[r] && n < besti[r])) {
        bestv[r] = sc; besti[r] = n;
      }
    }
  }

  // reduce across the 16 lanes holding the same token row (xor within half)
#pragma unroll
  for (int r = 0; r < 8; ++r) {
#pragma unroll
    for (int mask = 1; mask < 16; mask <<= 1) {
      float ov = __shfl_xor(bestv[r], mask, 32);
      int   oi = __shfl_xor(besti[r], mask, 32);
      if (ov < bestv[r] || (ov == bestv[r] && oi < besti[r])) {
        bestv[r] = ov; besti[r] = oi;
      }
    }
    if (ncol == 0) {
      int m = r + half * 8;
      int tok = tok0 + m;
      if (tok < tokens) {
        idx_out[tok] = besti[r];
        int b = tok / T, t = tok % T;
        codes_out[((long)b * NQ + q) * T + t] = (float)besti[r];
      }
    }
  }
}

// gather chosen code, update quantized/residual, accumulate sum((qs-r)^2)
__global__ void vq_apply_k(float* __restrict__ resid, float* __restrict__ quant,
                           const float* __restrict__ cb, const int* __restrict__ idx,
                           float* __restrict__ loss_acc, int tokens)
{
  int tok = blockIdx.x;
  if (tok >= tokens) return;
  const float* crow = cb + (long)idx[tok] * 1024;
  float* rrow = resid + (long)tok * 1024;
  float* qrow = quant + (long)tok * 1024;
  float ls = 0.f;
  for (int d = threadIdx.x; d < 1024; d += blockDim.x) {
    float qv = crow[d], rv = rrow[d];
    float df = qv - rv;
    ls += df * df;
    qrow[d] += qv;
    rrow[d]  = rv - qv;
  }
  __shared__ float sh[256];
  sh[threadIdx.x] = ls;
  __syncthreads();
  for (int o = 128; o > 0; o >>= 1) {
    if (threadIdx.x < o) sh[threadIdx.x] += sh[threadIdx.x + o];
    __syncthreads();
  }
  if (threadIdx.x == 0) atomicAdd(loss_acc, sh[0]);
}

// tanh, crop/pad to L, write rec to d_out, accumulate sum|rec-audio|
__global__ void tanh_store_l1_k(const float* __restrict__ pre,
                                const float* __restrict__ audio,
                                float* __restrict__ out_rec,
                                float* __restrict__ l1_acc,
                                int B, int Lr, int L)
{
  long n = (long)B * L;
  float ls = 0.f;
  for (long i = (long)blockIdx.x * blockDim.x + threadIdx.x; i < n;
       i += (long)gridDim.x * blockDim.x) {
    int b = (int)(i / L);
    int pos = (int)(i % L);
    float r = (pos < Lr) ? tanhf(pre[(long)b * Lr + pos]) : 0.f;
    out_rec[i] = r;
    ls += fabsf(r - audio[i]);
  }
  __shared__ float sh[256];
  sh[threadIdx.x] = ls;
  __syncthreads();
  for (int o = 128; o > 0; o >>= 1) {
    if (threadIdx.x < o) sh[threadIdx.x] += sh[threadIdx.x + o];
    __syncthreads();
  }
  if (threadIdx.x == 0) atomicAdd(l1_acc, sh[0]);
}

__global__ void finalize_k(const float* __restrict__ acc, float* __restrict__ out,
                           float nrec, float nvq)
{
  float recon = acc[0] / nrec;                 // mean |rec - audio|
  float vq    = 1.25f * acc[1] / (nvq * 8.f);  // (cb + 0.25*commit)/NQ, identical MSEs
  out[0] = recon + vq;
  out[1] = recon;
  out[2] = vq;
}

// ============================================================================
// Host-side orchestration
// ============================================================================
extern "C" void kernel_launch(void* const* d_in, const int* in_sizes, int n_in,
                              void* d_out, int out_size, void* d_ws, size_t ws_size,
                              hipStream_t stream)
{
  (void)in_sizes; (void)n_in; (void)out_size; (void)ws_size;
  const int Bn = 4, Laud = 160000, D = 1024, Tt = 1999, NQl = 8, NCl = 1024;
  const long TOK = (long)Bn * Tt;

  const float* audio = (const float*)d_in[0];
  auto F = [&](int i) { return (const float*)d_in[i]; };

  // ---- parameter index map (setup_inputs insertion order, nested dicts flat) ----
  int idx = 1;
  int proj_w = idx++, proj_b = idx++;
  int dw[4], db[4], erb[4];
  for (int i = 0; i < 4; ++i) { dw[i] = idx++; db[i] = idx++; erb[i] = idx; idx += 10; }
  int eout_w = idx++, eout_b = idx++;
  int dproj_w = idx++, dproj_b = idx++;
  int tw[5], tb[5], drb[4];
  for (int i = 0; i < 5; ++i) { tw[i] = idx++; tb[i] = idx++; if (i < 4) { drb[i] = idx; idx += 10; } }
  int fin_w = idx++, fin_b = idx++, cbks = idx++;

  // ---- workspace carve ----
  char* wsb = (char*)d_ws;
  size_t off = 0;
  auto carve = [&](size_t bytes) {
    void* p = wsb + off;
    off = (off + bytes + 255) & ~(size_t)255;
    return p;
  };
  const size_t BIGN = (size_t)32 * 160000 * Bn;   // largest activation (floats)
  float* bufA  = (float*)carve(BIGN * 4);
  float* bufB  = (float*)carve(BIGN * 4);
  float* bufC  = (float*)carve(BIGN * 4);
  float* resid = (float*)carve((size_t)TOK * D * 4);
  float* quant = (float*)carve((size_t)TOK * D * 4);
  float* zq    = (float*)carve((size_t)Bn * D * Tt * 4);
  float* csq   = (float*)carve((size_t)NQl * NCl * 4);
  float* stats = (float*)carve((size_t)Bn * 2 * 4);
  float* dacc  = (float*)carve(2 * 4);            // [0]=L1 sum, [1]=VQ sumsq
  int*   idxb  = (int*)carve((size_t)TOK * 4);

  float* outRec   = (float*)d_out;
  float* outCodes = outRec + (size_t)Bn * Laud;
  float* outLoss  = outCodes + (size_t)Bn * NQl * Tt;

  // ---- launch helpers ----
  auto conv = [&](const float* x, const float* w, const float* b, float* y,
                  int Cin, int Cout, int K, int s, int padL, int Lin, int Lout, int mode) {
    dim3 g((Lout + 127) / 128, (Cout + 15) / 16, Bn);
    dim3 blk(256);
#define LCONV(KW, ST, MD) \
    conv1d_wmma_k<KW, ST, MD><<<g, blk, 0, stream>>>(x, w, b, y, Cin, Cout, padL, Lin, Lout)
    if (mode == 0) {
      if      (K == 7  && s == 1) LCONV(7, 1, 0);
      else if (K == 4  && s == 2) LCONV(4, 2, 0);
      else if (K == 3  && s == 1) LCONV(3, 1, 0);
      else if (K == 1  && s == 1) LCONV(1, 1, 0);
      else if (K == 10 && s == 5) LCONV(10, 5, 0);
    } else {
      if      (K == 10 && s == 5) LCONV(10, 5, 1);
      else if (K == 4  && s == 2) LCONV(4, 2, 1);
    }
#undef LCONV
  };
  auto ew_grid = [&](long n) { return dim3((unsigned)((n + 255) / 256)); };
  auto resblock = [&](float* x, float* t1, float* t2, int C, int L, int pb) {
    long n = (long)Bn * C * L;
    gn_stats_k<<<Bn, 256, 0, stream>>>(x, stats, C, L);
    gn_snake_k<<<ew_grid(n), 256, 0, stream>>>(x, t1, stats, F(pb + 0), F(pb + 1),
                                               F(pb + 2), C, L, n);
    conv(t1, F(pb + 3), F(pb + 4), t2, C, C, 3, 1, 2, L, L, 0);   // causal k3
    gn_stats_k<<<Bn, 256, 0, stream>>>(t2, stats, C, L);
    gn_snake_k<<<ew_grid(n), 256, 0, stream>>>(t2, t1, stats, F(pb + 5), F(pb + 6),
                                               F(pb + 7), C, L, n);
    conv(t1, F(pb + 8), F(pb + 9), t2, C, C, 1, 1, 0, L, L, 0);   // 1x1
    add_inplace_k<<<ew_grid(n), 256, 0, stream>>>(x, t2, n);      // + residual
  };

  // ---- init accumulators ----
  fill_zero_k<<<1, 32, 0, stream>>>(dacc, 2);
  fill_zero_k<<<ew_grid(TOK * D), 256, 0, stream>>>(quant, TOK * D);

  // ======================= encoder =======================
  float *X = bufA, *Y = bufB, *Z = bufC;
  conv(audio, F(proj_w), F(proj_b), X, 1, 32, 7, 1, 3, Laud, Laud, 0);
  int chans[5] = {32, 64, 128, 256, 512};
  int C = 32, L = Laud;
  for (int i = 0; i < 4; ++i) {
    int Co = chans[i + 1], Lo = L / 2;
    conv(X, F(dw[i]), F(db[i]), Y, C, Co, 4, 2, 1, L, Lo, 0);
    resblock(Y, Z, X, Co, Lo, erb[i]);
    float* t = X; X = Y; Y = t;
    C = Co; L = Lo;
  }
  conv(X, F(eout_w), F(eout_b), Y, 512, 1024, 10, 5, 2, L, Tt, 0);  // z in Y

  // ======================= residual VQ =======================
  transpose_dt_k<<<ew_grid((long)Bn * D * Tt), 256, 0, stream>>>(Y, resid, Bn, D, Tt);
  csq_k<<<(NQl * NCl + 255) / 256, 256, 0, stream>>>(F(cbks), csq, NQl * NCl, D);
  for (int q = 0; q < NQl; ++q) {
    const float* cbq = F(cbks) + (size_t)q * NCl * D;
    vq_argmin_k<<<(unsigned)((TOK + 15) / 16), 32, 0, stream>>>(
        resid, cbq, csq + q * NCl, outCodes, idxb, (int)TOK, Tt, q, NQl);
    vq_apply_k<<<(unsigned)TOK, 256, 0, stream>>>(resid, quant, cbq, idxb,
                                                  dacc + 1, (int)TOK);
  }
  transpose_td_k<<<ew_grid((long)Bn * D * Tt), 256, 0, stream>>>(quant, zq, Bn, D, Tt);

  // ======================= decoder =======================
  conv(zq, F(dproj_w), F(dproj_b), X, 1024, 512, 7, 1, 3, Tt, Tt, 0);
  int dch[6] = {512, 256, 128, 64, 32, 32};
  int s_[5] = {5, 2, 2, 2, 2}, p_[5] = {2, 1, 1, 1, 1}, k_[5] = {10, 4, 4, 4, 4};
  C = 512; L = Tt;
  for (int i = 0; i < 5; ++i) {
    int Co = dch[i + 1];
    int Lo = (L - 1) * s_[i] + k_[i] - 2 * p_[i];
    conv(X, F(tw[i]), F(tb[i]), Y, C, Co, k_[i], s_[i], p_[i], L, Lo, 1); // convT
    if (i < 4) resblock(Y, Z, X, Co, Lo, drb[i]);
    float* t = X; X = Y; Y = t;
    C = Co; L = Lo;
  }
  conv(X, F(fin_w), F(fin_b), Y, 32, 1, 7, 1, 3, L, L, 0);  // pre-tanh in Y

  tanh_store_l1_k<<<1024, 256, 0, stream>>>(Y, audio, outRec, dacc, Bn, L, Laud);
  finalize_k<<<1, 1, 0, stream>>>(dacc, outLoss, (float)((long)Bn * Laud),
                                  (float)(TOK * (long)D));
}
